// PointPillars_86517821213346
// MI455X (gfx1250) — matrix operand order repaired and checked by reading, not compiled, not run on hardware
//
#include <hip/hip_runtime.h>
#include <hip/hip_fp16.h>
#include <stdint.h>

// ---------------------------------------------------------------------------
// CDNA5 (gfx1250) PointPillars forward.  wave32 WMMA f16->f32 implicit GEMM.
// All feature maps are NHWC f16 with a 1-pixel zero halo so the 3x3 conv
// inner loop is completely branch-free (no EXEC juggling, no zero-fills).
// Each wave computes a 32x32 output tile with 4 independent accumulators.
// ---------------------------------------------------------------------------

typedef __attribute__((ext_vector_type(16))) _Float16 v16h;
typedef __attribute__((ext_vector_type(8)))  float    v8f;

union Frag32 {            // 32 bytes = one WMMA 16-bit A/B fragment per lane
  float4 q[2];
  v16h   v;
};

__device__ __forceinline__ void frag_load(Frag32& f, const _Float16* p0,
                                          const _Float16* p1) {
  f.q[0] = *reinterpret_cast<const float4*>(p0);
  f.q[1] = *reinterpret_cast<const float4*>(p1);
}

// ---------------------------------------------------------------------------
// Utility: zero fill / halo zero
// ---------------------------------------------------------------------------
__global__ void fill_zero_u32(uint32_t* __restrict__ p, long long n) {
  long long i = (long long)blockIdx.x * blockDim.x + threadIdx.x;
  const long long stride = (long long)gridDim.x * blockDim.x;
  for (; i < n; i += stride) p[i] = 0u;
}

// zero the 1-pixel border of a padded NHWC (2,Hp,Wp,C) buffer
__global__ void halo_zero(_Float16* __restrict__ buf, int Hp, int Wp, int C) {
  const int pix = blockIdx.x * blockDim.x + threadIdx.x;
  const int tot = 2 * Hp * Wp;
  if (pix >= tot) return;
  const int b = pix / (Hp * Wp);
  const int r = pix - b * (Hp * Wp);
  const int y = r / Wp;
  const int x = r - y * Wp;
  if (y == 0 || y == Hp - 1 || x == 0 || x == Wp - 1) {
    float4 z; z.x = 0.f; z.y = 0.f; z.z = 0.f; z.w = 0.f;
    _Float16* d = buf + (size_t)pix * C;
    for (int c = 0; c < C; c += 8)
      *reinterpret_cast<float4*>(d + c) = z;
  }
}

// ---------------------------------------------------------------------------
// PFN: per-pillar 9->64 MLP + BN + ReLU + max over 32 points, scattered into
// the padded f16 NHWC canvas (2 x 498 x 434 x 64), interior at +1.
// ---------------------------------------------------------------------------
#define CAN_HP 498
#define CAN_WP 434

__global__ void pfn_scatter_kernel(const float* __restrict__ pillars,  // (P,32,4)
                                   const int*   __restrict__ coors,    // (P,4)
                                   const int*   __restrict__ npoints,  // (P)
                                   const float* __restrict__ w,        // (64,9)
                                   const float* __restrict__ g,        // (64)
                                   const float* __restrict__ bb,       // (64)
                                   _Float16*    __restrict__ canvas) {
  const int p = blockIdx.x;
  const int t = threadIdx.x;            // 64 threads = 64 output channels
  __shared__ float sp[32][4];
  __shared__ float ctr[3];
  const int np = npoints[p];
  if (t < 32) {
    const float* src = pillars + ((size_t)p * 32 + t) * 4;
    sp[t][0] = src[0]; sp[t][1] = src[1]; sp[t][2] = src[2]; sp[t][3] = src[3];
  }
  __syncthreads();
  if (t < 3) {
    float s = 0.f;
    for (int j = 0; j < 32; ++j) if (j < np) s += sp[j][t];
    ctr[t] = s / (float)np;
  }
  __syncthreads();

  float wv[9];
#pragma unroll
  for (int c = 0; c < 9; ++c) wv[c] = w[t * 9 + c];
  const float gg = g[t], bbv = bb[t];

  const int bidx = coors[p * 4 + 0];
  const int xi   = coors[p * 4 + 1];
  const int yi   = coors[p * 4 + 2];
  const float cx = xi * 0.16f + 0.08f;
  const float cy = yi * 0.16f - 39.6f;

  float mx = -1e30f;
  for (int j = 0; j < 32; ++j) {
    float dot = 0.f;
    if (j < np) {
      const float x = sp[j][0], y = sp[j][1], z = sp[j][2], r = sp[j][3];
      const float f0 = x - cx, f1 = y - cy;
      const float f4 = x - ctr[0], f5 = y - ctr[1], f6 = z - ctr[2];
      dot = f0 * wv[0] + f1 * wv[1] + z * wv[2] + r * wv[3] +
            f4 * wv[4] + f5 * wv[5] + f6 * wv[6] + f0 * wv[7] + f1 * wv[8];
    }
    float h = dot * gg + bbv;
    h = h > 0.f ? h : 0.f;
    mx = h > mx ? h : mx;
  }
  canvas[(((size_t)bidx * CAN_HP + (yi + 1)) * CAN_WP + (xi + 1)) * 64 + t] =
      (_Float16)mx;
}

// ---------------------------------------------------------------------------
// Weight repack kernels (fold BN gamma into f16 weights, copy beta as bias)
// ---------------------------------------------------------------------------
__global__ void repack_conv_w(const float* __restrict__ w,   // OIHW (K,C,3,3)
                              const float* __restrict__ g,
                              const float* __restrict__ b,
                              _Float16* __restrict__ wdst,   // (K, 9, C)
                              float* __restrict__ bdst, int K, int C) {
  const int t = blockIdx.x * blockDim.x + threadIdx.x;
  const int total = K * 9 * C;
  if (t < total) {
    const int o   = t / (9 * C);
    const int r   = t - o * 9 * C;
    const int tap = r / C;
    const int c   = r - tap * C;
    wdst[t] = (_Float16)(w[((size_t)o * C + c) * 9 + tap] * g[o]);
  }
  if (t < K) bdst[t] = b[t];
}

__global__ void repack_up_w(const float* __restrict__ w,   // (Cin, 128, s, s)
                            const float* __restrict__ g,
                            const float* __restrict__ b,
                            _Float16* __restrict__ wdst,   // (ss, 128, Cin)
                            float* __restrict__ bdst, int Cin, int ss) {
  const int t = blockIdx.x * blockDim.x + threadIdx.x;
  const int total = ss * 128 * Cin;
  if (t < total) {
    const int st = t / (128 * Cin);
    const int r  = t - st * 128 * Cin;
    const int o  = r / Cin;
    const int c  = r - o * Cin;
    wdst[t] = (_Float16)(w[((size_t)c * 128 + o) * ss + st] * g[o]);
  }
  if (t < 128) bdst[t] = b[t];
}

__global__ void repack_head_w(const float* __restrict__ cw, const float* __restrict__ cb,
                              const float* __restrict__ rw, const float* __restrict__ rb,
                              const float* __restrict__ dw, const float* __restrict__ db,
                              _Float16* __restrict__ wdst,  // (80, 384)
                              float* __restrict__ bdst) {   // (80)
  const int t = blockIdx.x * blockDim.x + threadIdx.x;
  if (t < 80 * 384) {
    const int o = t / 384, c = t - o * 384;
    float v = 0.f;
    if (o < 18)      v = cw[o * 384 + c];
    else if (o < 60) v = rw[(o - 18) * 384 + c];
    else if (o < 72) v = dw[(o - 60) * 384 + c];
    wdst[t] = (_Float16)v;
  }
  if (t < 80) {
    float v = 0.f;
    if (t < 18)      v = cb[t];
    else if (t < 60) v = rb[t - 18];
    else if (t < 72) v = db[t - 60];
    bdst[t] = v;
  }
}

// ---------------------------------------------------------------------------
// 3x3 conv on halo-padded NHWC maps, implicit GEMM via WMMA 16x16x32 f16.
// One wave -> 32 pixels x 32 out-channels (4 accumulators, 2x2 sub-tiles).
// Inner loop is branch-free: 4 fragment loads -> 4 v_wmma per k-slice.
// in : (2, Hp,  Wp,  CIN)  padded, interior at +1
// out: (2, Hop, Wop, Cout) padded, interior at +1
// ---------------------------------------------------------------------------
template <int CIN>
__global__ void conv3x3_wmma(const _Float16* __restrict__ in,
                             const _Float16* __restrict__ wgt,   // (Cout,9,CIN)
                             const float* __restrict__ bias,
                             _Float16* __restrict__ out,
                             int Hp, int Wp, int Hout, int Wout,
                             int Cout, int stride, int mtiles, int ntiles) {
  const int wave = blockIdx.x * (blockDim.x >> 5) + (threadIdx.x >> 5);
  if (wave >= mtiles * ntiles) return;
  const int mt   = wave / ntiles;
  const int nt   = wave - mt * ntiles;
  const int lane = threadIdx.x & 31;
  const int hsel = lane >> 4;
  const int mrow = lane & 15;
  const int n0   = nt * 32;
  const int HWo  = Hout * Wout;
  const int Mtot = 2 * HWo;
  const int m0   = mt * 32;
  const int Hop  = Hout + 2, Wop = Wout + 2;

  // pixel bases for the two 16-row A sub-tiles (clamped for partial tiles)
  const _Float16* abase[2];
#pragma unroll
  for (int i = 0; i < 2; ++i) {
    int m = m0 + 16 * i + mrow;
    if (m >= Mtot) m = Mtot - 1;
    const int b  = m / HWo;
    const int pr = m - b * HWo;
    const int oy = pr / Wout;
    const int ox = pr - oy * Wout;
    abase[i] = in + ((size_t)(b * Hp + oy * stride) * Wp + ox * stride) * CIN;
  }

  const _Float16* wrow0 = wgt + (size_t)(n0 + mrow) * 9 * CIN;
  const _Float16* wrow1 = wgt + (size_t)(n0 + 16 + mrow) * 9 * CIN;

  v8f acc00, acc01, acc10, acc11;
#pragma unroll
  for (int i = 0; i < 8; ++i) { acc00[i] = 0.f; acc01[i] = 0.f;
                                acc10[i] = 0.f; acc11[i] = 0.f; }

  for (int tap = 0; tap < 9; ++tap) {
    const int ky = tap / 3, kx = tap - ky * 3;
    const _Float16* a0 = abase[0] + ((size_t)ky * Wp + kx) * CIN;
    const _Float16* a1 = abase[1] + ((size_t)ky * Wp + kx) * CIN;
    const _Float16* w0 = wrow0 + tap * CIN;
    const _Float16* w1 = wrow1 + tap * CIN;
    __builtin_prefetch(w0 + CIN, 0, 3);   // global_prefetch_b8: next tap weights
#pragma unroll
    for (int cc = 0; cc < CIN; cc += 32) {
      Frag32 A0, A1, B0, B1;
      frag_load(A0, a0 + cc + hsel * 8,  a0 + cc + 16 + hsel * 8);
      frag_load(A1, a1 + cc + hsel * 8,  a1 + cc + 16 + hsel * 8);
      frag_load(B0, w0 + cc + hsel * 8,  w0 + cc + 16 + hsel * 8);
      frag_load(B1, w1 + cc + hsel * 8,  w1 + cc + 16 + hsel * 8);
      acc00 = __builtin_amdgcn_wmma_f32_16x16x32_f16(false, A0.v, false, B0.v,
                                                     (short)0, acc00, false, false);
      acc01 = __builtin_amdgcn_wmma_f32_16x16x32_f16(false, A0.v, false, B1.v,
                                                     (short)0, acc01, false, false);
      acc10 = __builtin_amdgcn_wmma_f32_16x16x32_f16(false, A1.v, false, B0.v,
                                                     (short)0, acc10, false, false);
      acc11 = __builtin_amdgcn_wmma_f32_16x16x32_f16(false, A1.v, false, B1.v,
                                                     (short)0, acc11, false, false);
    }
  }

  const int   ch0 = n0 + mrow, ch1 = ch0 + 16;
  const float bv0 = bias[ch0], bv1 = bias[ch1];
#pragma unroll
  for (int i = 0; i < 2; ++i) {
    const v8f& aN0 = (i == 0) ? acc00 : acc10;
    const v8f& aN1 = (i == 0) ? acc01 : acc11;
#pragma unroll
    for (int r = 0; r < 8; ++r) {
      const int mo = m0 + 16 * i + r + hsel * 8;
      if (mo < Mtot) {
        const int b  = mo / HWo;
        const int pr = mo - b * HWo;
        const int oy = pr / Wout;
        const int ox = pr - oy * Wout;
        _Float16* d = out + ((size_t)(b * Hop + oy + 1) * Wop + ox + 1) * Cout;
        float v0 = aN0[r] + bv0; v0 = v0 > 0.f ? v0 : 0.f;
        float v1 = aN1[r] + bv1; v1 = v1 > 0.f ? v1 : 0.f;
        d[ch0] = (_Float16)v0;
        d[ch1] = (_Float16)v1;
      }
    }
  }
}

// ---------------------------------------------------------------------------
// Neck: ConvTranspose (kernel==stride) == ss independent 1x1 GEMMs.
// Reads padded input interior, writes BN+ReLU into unpadded (2,248,216,384).
// ---------------------------------------------------------------------------
template <int CIN>
__global__ void upconv_wmma(const _Float16* __restrict__ in,    // (2,Hp,Wp,CIN) padded
                            const _Float16* __restrict__ wgt,   // (ss,128,CIN)
                            const float* __restrict__ bias,     // (128)
                            _Float16* __restrict__ fout,        // (2,248,216,384)
                            int Hin, int Win, int Hp, int Wp,
                            int scale, int chan_off, int mtiles) {
  const int st  = blockIdx.y;
  const int s_i = st / scale;
  const int t_i = st - s_i * scale;
  const int ntiles = 8;  // 128 out channels
  const int wave = blockIdx.x * (blockDim.x >> 5) + (threadIdx.x >> 5);
  if (wave >= mtiles * ntiles) return;
  const int mt   = wave / ntiles;
  const int nt   = wave - mt * ntiles;
  const int lane = threadIdx.x & 31;
  const int hsel = lane >> 4;
  const int mrow = lane & 15;
  const int n0   = nt * 16;
  const int HWi  = Hin * Win;
  const int Mtot = 2 * HWi;

  int m = mt * 16 + mrow;
  if (m >= Mtot) m = Mtot - 1;
  const int b  = m / HWi;
  const int pr = m - b * HWi;
  const int hh = pr / Win;
  const int ww = pr - hh * Win;

  v8f acc;
#pragma unroll
  for (int i = 0; i < 8; ++i) acc[i] = 0.f;

  const _Float16* arow = in + ((size_t)(b * Hp + hh + 1) * Wp + ww + 1) * CIN;
  const _Float16* wrow = wgt + (size_t)(st * 128 + n0 + mrow) * CIN;
#pragma unroll
  for (int cc = 0; cc < CIN; cc += 32) {
    Frag32 A, B;
    frag_load(A, arow + cc + hsel * 8,  arow + cc + 16 + hsel * 8);
    frag_load(B, wrow + cc + hsel * 8,  wrow + cc + 16 + hsel * 8);
    acc = __builtin_amdgcn_wmma_f32_16x16x32_f16(false, A.v, false, B.v,
                                                 (short)0, acc, false, false);
  }
  const float bv = bias[n0 + mrow];
#pragma unroll
  for (int r = 0; r < 8; ++r) {
    const int mo = mt * 16 + r + hsel * 8;
    if (mo < Mtot) {
      const int bb = mo / HWi;
      const int pp = mo - bb * HWi;
      const int h2 = pp / Win;
      const int w2 = pp - h2 * Win;
      const int oy = h2 * scale + s_i;
      const int ox = w2 * scale + t_i;
      float v = acc[r] + bv;
      v = v > 0.f ? v : 0.f;
      fout[(((size_t)bb * 248 + oy) * 216 + ox) * 384 + chan_off + (n0 + mrow)] =
          (_Float16)v;
    }
  }
}

// ---------------------------------------------------------------------------
// Head: fused 1x1 conv 384 -> 80 (72 used: 18 cls | 42 reg | 12 dir),
// WMMA GEMM, f32 NCHW output straight into d_out.
// ---------------------------------------------------------------------------
#define HD_HW   53568              // 248*216
#define HD_M    107136             // 2*HD_HW
#define BASE_REG 1928448           // 2*18*HD_HW
#define BASE_DIR 6428160           // BASE_REG + 2*42*HD_HW

__global__ void head_wmma(const _Float16* __restrict__ f,   // (HD_M, 384)
                          const _Float16* __restrict__ hw,  // (80, 384)
                          const float* __restrict__ hb,     // (80)
                          float* __restrict__ out,
                          int mtiles) {
  const int ntiles = 5;  // 80 padded out channels
  const int wave = blockIdx.x * (blockDim.x >> 5) + (threadIdx.x >> 5);
  if (wave >= mtiles * ntiles) return;
  const int mt   = wave / ntiles;
  const int nt   = wave - mt * ntiles;
  const int lane = threadIdx.x & 31;
  const int hsel = lane >> 4;
  const int mrow = lane & 15;
  const int n0   = nt * 16;
  const int m    = mt * 16 + mrow;     // HD_M divisible by 16

  v8f acc;
#pragma unroll
  for (int i = 0; i < 8; ++i) acc[i] = 0.f;

  const _Float16* arow = f + (size_t)m * 384;
  const _Float16* wrow = hw + (size_t)(n0 + mrow) * 384;
#pragma unroll
  for (int cc = 0; cc < 384; cc += 32) {
    Frag32 A, B;
    frag_load(A, arow + cc + hsel * 8,  arow + cc + 16 + hsel * 8);
    frag_load(B, wrow + cc + hsel * 8,  wrow + cc + 16 + hsel * 8);
    acc = __builtin_amdgcn_wmma_f32_16x16x32_f16(false, A.v, false, B.v,
                                                 (short)0, acc, false, false);
  }
  const int ch = n0 + mrow;
  if (ch >= 72) return;
  const float bv = hb[ch];
#pragma unroll
  for (int r = 0; r < 8; ++r) {
    const int mo = mt * 16 + r + hsel * 8;
    const int b  = mo / HD_HW;
    const int px = mo - b * HD_HW;
    const float v = acc[r] + bv;
    size_t idx;
    if (ch < 18)      idx = ((size_t)b * 18 + ch) * HD_HW + px;
    else if (ch < 60) idx = BASE_REG + ((size_t)b * 42 + (ch - 18)) * HD_HW + px;
    else              idx = BASE_DIR + ((size_t)b * 12 + (ch - 60)) * HD_HW + px;
    out[idx] = v;
  }
}

// ---------------------------------------------------------------------------
// Host orchestration
// ---------------------------------------------------------------------------
extern "C" void kernel_launch(void* const* d_in, const int* in_sizes, int n_in,
                              void* d_out, int out_size, void* d_ws, size_t ws_size,
                              hipStream_t stream) {
  const float* pillars = (const float*)d_in[0];
  const int*   coors   = (const int*)d_in[1];
  const int*   npts    = (const int*)d_in[2];
  const float* pfn_w   = (const float*)d_in[3];
  const float* pfn_g   = (const float*)d_in[4];
  const float* pfn_b   = (const float*)d_in[5];
  const float* b1_w0   = (const float*)d_in[6];
  const float* b1_w    = (const float*)d_in[7];
  const float* b1_g    = (const float*)d_in[8];
  const float* b1_b    = (const float*)d_in[9];
  const float* b2_w0   = (const float*)d_in[10];
  const float* b2_w    = (const float*)d_in[11];
  const float* b2_g    = (const float*)d_in[12];
  const float* b2_b    = (const float*)d_in[13];
  const float* b3_w0   = (const float*)d_in[14];
  const float* b3_w    = (const float*)d_in[15];
  const float* b3_g    = (const float*)d_in[16];
  const float* b3_b    = (const float*)d_in[17];
  const float* n1_w    = (const float*)d_in[18];
  const float* n2_w    = (const float*)d_in[19];
  const float* n3_w    = (const float*)d_in[20];
  const float* n_g     = (const float*)d_in[21];
  const float* n_b     = (const float*)d_in[22];
  const float* cls_w   = (const float*)d_in[23];
  const float* cls_b   = (const float*)d_in[24];
  const float* reg_w   = (const float*)d_in[25];
  const float* reg_b   = (const float*)d_in[26];
  const float* dir_w   = (const float*)d_in[27];
  const float* dir_b   = (const float*)d_in[28];

  // ---- workspace layout (bytes) ----
  auto align256 = [](size_t x) { return (x + 255) & ~(size_t)255; };
  const size_t SZ_CANVAS = (size_t)2 * CAN_HP * CAN_WP * 64 * 2;   // padded
  const size_t SZ_P      = (size_t)2 * 250 * 218 * 64 * 2;          // max padded map
  const size_t SZ_F      = (size_t)2 * 248 * 216 * 384 * 2;
  const size_t SZ_WBUF   = (size_t)256 * 9 * 256 * 2;
  const size_t SZ_NW     = (size_t)16 * 128 * 256 * 2;

  char* ws = (char*)d_ws;
  size_t off = 0;
  _Float16* canvas = (_Float16*)(ws + off); off = align256(off + SZ_CANVAS);
  _Float16* p0     = (_Float16*)(ws + off); off = align256(off + SZ_P);
  _Float16* p1     = (_Float16*)(ws + off); off = align256(off + SZ_P);
  _Float16* p2     = (_Float16*)(ws + off); off = align256(off + SZ_P);
  _Float16* p3     = (_Float16*)(ws + off); off = align256(off + SZ_P);
  _Float16* fbuf   = (_Float16*)(ws + off); off = align256(off + SZ_F);
  _Float16* wbuf   = (_Float16*)(ws + off); off = align256(off + SZ_WBUF);
  float*    bbuf   = (float*)(ws + off);    off = align256(off + 256 * sizeof(float));
  _Float16* nwbuf  = (_Float16*)(ws + off); off = align256(off + SZ_NW);
  float*    nbbuf  = (float*)(ws + off);    off = align256(off + 128 * sizeof(float));
  _Float16* hwbuf  = (_Float16*)(ws + off); off = align256(off + (size_t)80 * 384 * 2);
  float*    hbbuf  = (float*)(ws + off);    off = align256(off + 80 * sizeof(float));
  (void)ws_size; (void)in_sizes; (void)n_in; (void)out_size;

  // ---- 1) canvas = 0 (covers halo), PFN + scatter ----
  fill_zero_u32<<<2048, 256, 0, stream>>>((uint32_t*)canvas,
                                          (long long)(SZ_CANVAS / 4));
  pfn_scatter_kernel<<<24000, 64, 0, stream>>>(pillars, coors, npts,
                                               pfn_w, pfn_g, pfn_b, canvas);

  // ---- 2) backbone convs (halo-padded, template dispatch on Cin) ----
  auto conv = [&](const _Float16* in, _Float16* outp,
                  const float* w, const float* g, const float* b,
                  int Hp, int Wp, int Hout, int Wout,
                  int Cin, int Cout, int stride) {
    const int total = Cout * 9 * Cin;
    repack_conv_w<<<(total + 255) / 256, 256, 0, stream>>>(w, g, b, wbuf, bbuf,
                                                           Cout, Cin);
    const int Mtot   = 2 * Hout * Wout;
    const int mtiles = (Mtot + 31) / 32;
    const int ntiles = Cout / 32;
    const int waves  = mtiles * ntiles;
    const int blocks = (waves + 3) / 4;
    if (Cin == 64)
      conv3x3_wmma<64><<<blocks, 128, 0, stream>>>(in, wbuf, bbuf, outp, Hp, Wp,
                                                   Hout, Wout, Cout, stride,
                                                   mtiles, ntiles);
    else if (Cin == 128)
      conv3x3_wmma<128><<<blocks, 128, 0, stream>>>(in, wbuf, bbuf, outp, Hp, Wp,
                                                    Hout, Wout, Cout, stride,
                                                    mtiles, ntiles);
    else
      conv3x3_wmma<256><<<blocks, 128, 0, stream>>>(in, wbuf, bbuf, outp, Hp, Wp,
                                                    Hout, Wout, Cout, stride,
                                                    mtiles, ntiles);
  };
  auto halo = [&](_Float16* buf, int Hp, int Wp, int C) {
    const int pix = 2 * Hp * Wp;
    halo_zero<<<(pix + 255) / 256, 256, 0, stream>>>(buf, Hp, Wp, C);
  };

  // Block 1: 64ch, canvas(498x434) -> 250x218-padded 248x216 maps
  halo(p0, 250, 218, 64);
  halo(p1, 250, 218, 64);
  conv(canvas, p0, b1_w0,               b1_g + 0 * 64, b1_b + 0 * 64, 498, 434, 248, 216, 64, 64, 2);
  conv(p0, p1, b1_w + 0 * 64 * 64 * 9, b1_g + 1 * 64, b1_b + 1 * 64, 250, 218, 248, 216, 64, 64, 1);
  conv(p1, p0, b1_w + 1 * 64 * 64 * 9, b1_g + 2 * 64, b1_b + 2 * 64, 250, 218, 248, 216, 64, 64, 1);
  conv(p0, p1, b1_w + 2 * 64 * 64 * 9, b1_g + 3 * 64, b1_b + 3 * 64, 250, 218, 248, 216, 64, 64, 1);
  // x1 = p1 (250x218x64 padded)

  // Block 2: 64->128, -> 126x110-padded 124x108 maps
  halo(p2, 126, 110, 128);
  halo(p3, 126, 110, 128);
  conv(p1, p2, b2_w0,                    b2_g + 0 * 128, b2_b + 0 * 128, 250, 218, 124, 108, 64, 128, 2);
  conv(p2, p3, b2_w + 0 * 128 * 128 * 9, b2_g + 1 * 128, b2_b + 1 * 128, 126, 110, 124, 108, 128, 128, 1);
  conv(p3, p2, b2_w + 1 * 128 * 128 * 9, b2_g + 2 * 128, b2_b + 2 * 128, 126, 110, 124, 108, 128, 128, 1);
  conv(p2, p3, b2_w + 2 * 128 * 128 * 9, b2_g + 3 * 128, b2_b + 3 * 128, 126, 110, 124, 108, 128, 128, 1);
  conv(p3, p2, b2_w + 3 * 128 * 128 * 9, b2_g + 4 * 128, b2_b + 4 * 128, 126, 110, 124, 108, 128, 128, 1);
  conv(p2, p3, b2_w + 4 * 128 * 128 * 9, b2_g + 5 * 128, b2_b + 5 * 128, 126, 110, 124, 108, 128, 128, 1);
  // x2 = p3 (126x110x128 padded)

  // Block 3: 128->256, -> 64x56-padded 62x54 maps
  halo(p0, 64, 56, 256);
  halo(p2, 64, 56, 256);
  conv(p3, p0, b3_w0,                    b3_g + 0 * 256, b3_b + 0 * 256, 126, 110, 62, 54, 128, 256, 2);
  conv(p0, p2, b3_w + 0 * 256 * 256 * 9, b3_g + 1 * 256, b3_b + 1 * 256, 64, 56, 62, 54, 256, 256, 1);
  conv(p2, p0, b3_w + 1 * 256 * 256 * 9, b3_g + 2 * 256, b3_b + 2 * 256, 64, 56, 62, 54, 256, 256, 1);
  conv(p0, p2, b3_w + 2 * 256 * 256 * 9, b3_g + 3 * 256, b3_b + 3 * 256, 64, 56, 62, 54, 256, 256, 1);
  conv(p2, p0, b3_w + 3 * 256 * 256 * 9, b3_g + 4 * 256, b3_b + 4 * 256, 64, 56, 62, 54, 256, 256, 1);
  conv(p0, p2, b3_w + 4 * 256 * 256 * 9, b3_g + 5 * 256, b3_b + 5 * 256, 64, 56, 62, 54, 256, 256, 1);
  // x3 = p2 (64x56x256 padded)

  // ---- 3) neck (transposed convs -> 384-ch concat buffer) ----
  auto up = [&](const _Float16* in, int Hin, int Win, int Hp, int Wp, int Cin,
                int scale, int chan_off,
                const float* w, const float* g, const float* b) {
    const int ss = scale * scale;
    const int total = ss * 128 * Cin;
    repack_up_w<<<(total + 255) / 256, 256, 0, stream>>>(w, g, b, nwbuf, nbbuf,
                                                         Cin, ss);
    const int Mtot   = 2 * Hin * Win;
    const int mtiles = (Mtot + 15) / 16;
    const int waves  = mtiles * 8;
    dim3 grid((waves + 3) / 4, ss);
    if (Cin == 64)
      upconv_wmma<64><<<grid, 128, 0, stream>>>(in, nwbuf, nbbuf, fbuf, Hin, Win,
                                                Hp, Wp, scale, chan_off, mtiles);
    else if (Cin == 128)
      upconv_wmma<128><<<grid, 128, 0, stream>>>(in, nwbuf, nbbuf, fbuf, Hin, Win,
                                                 Hp, Wp, scale, chan_off, mtiles);
    else
      upconv_wmma<256><<<grid, 128, 0, stream>>>(in, nwbuf, nbbuf, fbuf, Hin, Win,
                                                 Hp, Wp, scale, chan_off, mtiles);
  };
  up(p1, 248, 216, 250, 218, 64, 1, 0,    n1_w, n_g + 0 * 128, n_b + 0 * 128);
  up(p3, 124, 108, 126, 110, 128, 2, 128, n2_w, n_g + 1 * 128, n_b + 1 * 128);
  up(p2, 62, 54, 64, 56, 256, 4, 256,     n3_w, n_g + 2 * 128, n_b + 2 * 128);

  // ---- 4) head ----
  repack_head_w<<<(80 * 384 + 255) / 256, 256, 0, stream>>>(
      cls_w, cls_b, reg_w, reg_b, dir_w, dir_b, hwbuf, hbbuf);
  const int mtiles = HD_M / 16;          // 6696
  const int waves  = mtiles * 5;
  head_wmma<<<(waves + 3) / 4, 128, 0, stream>>>(fbuf, hwbuf, hbbuf,
                                                 (float*)d_out, mtiles);
}
